// TransformerBlock_55542517071983
// MI455X (gfx1250) — compile-verified, matrix operand order checked by reading, not run on hardware
//
#include <hip/hip_runtime.h>
#include <math.h>

// ---------------------------------------------------------------------------
// Types for CDNA5 WMMA (wave32, 16x16x32 bf16 -> f32)
// ---------------------------------------------------------------------------
typedef __attribute__((ext_vector_type(16))) __bf16 v16bf;
typedef __attribute__((ext_vector_type(8)))  float  v8f;

#define D_MODEL 1024
#define N_HEADS 16
#define D_HEAD  64
#define N_EXPERTS 8
#define EXPERT_SIZE 4096
#define N_TOK   8192          // B*S = 4*2048
#define SEQ     2048
#define N_SLOTS 16384         // N_TOK * TOP_K
#define F_CHUNK 1024

static __device__ __forceinline__ unsigned short f2bf(float f) {
  union { float f; unsigned u; } v; v.f = f;
  unsigned u = v.u;
  unsigned r = (u + 0x7FFFu + ((u >> 16) & 1u)) >> 16;
  return (unsigned short)r;
}
static __device__ __forceinline__ float bf2f(unsigned short h) {
  union { unsigned u; float f; } v; v.u = ((unsigned)h) << 16;
  return v.f;
}

// Load one A/B fragment (16 bf16) for v_wmma_f32_16x16x32_bf16.
// Layout (ISA 7.12.2): elems 0..7 = K {kh*8..kh*8+7}, 8..15 = K {16+kh*8..}.
static __device__ __forceinline__ v16bf load_frag(const unsigned short* p, int kh) {
  union { uint4 u[2]; v16bf v; } r;
  r.u[0] = *(const uint4*)(p + kh * 8);
  r.u[1] = *(const uint4*)(p + 16 + kh * 8);
  return r.v;
}

static __device__ __forceinline__ v8f wmma_bf16(v16bf a, v16bf b, v8f c) {
  return __builtin_amdgcn_wmma_f32_16x16x32_bf16(false, a, false, b, (short)0, c,
                                                 false, false);
}

// ---------------------------------------------------------------------------
// fp32 -> bf16 convert WITH transpose: src [R][C] f32  ->  dst [C][R] bf16.
// Tiled 32x32 through LDS; grid (C/32, R/32, nmat).
// ---------------------------------------------------------------------------
__global__ __launch_bounds__(256)
void transpose_convert_kernel(const float* __restrict__ src,
                              unsigned short* __restrict__ dst, int R, int C) {
  long mb = (long)blockIdx.z * R * C;
  src += mb;
  dst += mb;
  int r0 = blockIdx.y * 32, c0 = blockIdx.x * 32;
  __shared__ unsigned short tile[32][33];
  int tx = threadIdx.x & 31, ty0 = threadIdx.x >> 5;
#pragma unroll
  for (int j = 0; j < 4; ++j) {
    int ty = ty0 + j * 8;
    tile[ty][tx] = f2bf(src[(long)(r0 + ty) * C + c0 + tx]);
  }
  __syncthreads();
#pragma unroll
  for (int j = 0; j < 4; ++j) {
    int ty = ty0 + j * 8;
    dst[(long)(c0 + ty) * R + r0 + tx] = tile[tx][ty];
  }
}

// ---------------------------------------------------------------------------
// LayerNorm: one block (256 thr) per token, writes bf16
// ---------------------------------------------------------------------------
__global__ __launch_bounds__(256)
void layernorm_kernel(const float* __restrict__ x, const float* __restrict__ g,
                      const float* __restrict__ b, unsigned short* __restrict__ out) {
  int t = blockIdx.x;
  int tid = threadIdx.x;
  const float* xr = x + (long)t * D_MODEL;
  float vals[4];
  float s = 0.f, s2 = 0.f;
#pragma unroll
  for (int j = 0; j < 4; ++j) {
    vals[j] = xr[tid + j * 256];
    s += vals[j];
    s2 += vals[j] * vals[j];
  }
#pragma unroll
  for (int off = 16; off >= 1; off >>= 1) {
    s  += __shfl_xor(s,  off, 32);
    s2 += __shfl_xor(s2, off, 32);
  }
  __shared__ float ps[8], ps2[8];
  int lane = tid & 31, wid = tid >> 5;
  if (lane == 0) { ps[wid] = s; ps2[wid] = s2; }
  __syncthreads();
  float ts = 0.f, ts2 = 0.f;
#pragma unroll
  for (int i = 0; i < 8; ++i) { ts += ps[i]; ts2 += ps2[i]; }
  float mean = ts * (1.0f / D_MODEL);
  float var  = ts2 * (1.0f / D_MODEL) - mean * mean;
  float rstd = rsqrtf(var + 1e-5f);
#pragma unroll
  for (int j = 0; j < 4; ++j) {
    int d = tid + j * 256;
    float y = (vals[j] - mean) * rstd * g[d] + b[d];
    out[(long)t * D_MODEL + d] = f2bf(y);
  }
}

// ---------------------------------------------------------------------------
// Tiled WMMA GEMM, 128x128 block, K step 32, 256 threads (8 waves), LDS
// double-buffered & software-pipelined. B is pre-transposed: Bt[n][k].
// MODE 0: out = bf16(A@B + bias)                        (Q,K projections)
// MODE 1: out = f32 (A@B + bias + residual)             (O projection + skip)
// MODE 2: out = bf16(gelu(A@B + bias)), gathered rows   (MoE up)
// MODE 3: out f32: A@B + (first ? bias : out), gathered (MoE down, accum)
// MODE 4: out = bf16(A@B + bias) scattered to [bh][d][seq] (V projection)
// ---------------------------------------------------------------------------
template <int MODE>
__global__ __launch_bounds__(256)
void gemm_bf16_kernel(const unsigned short* __restrict__ A, int lda,
                      const unsigned short* __restrict__ Bt, int ldb,
                      long b_expert_stride,
                      const float* __restrict__ bias, int bias_stride,
                      void* __restrict__ Out, int ldo,
                      const float* __restrict__ Res,
                      int M, int N, int K,
                      const int* __restrict__ slot_list,
                      const int* __restrict__ counts,
                      int first_chunk) {
  constexpr bool MOE = (MODE == 2 || MODE == 3);
  __shared__ __align__(16) unsigned short As[2][128 * 40];
  __shared__ __align__(16) unsigned short Bs[2][128 * 40];
  __shared__ int a_rows[128];
  __shared__ int o_rows[128];

  int z  = blockIdx.z;
  int m0 = blockIdx.y * 128;
  int n0 = blockIdx.x * 128;
  int mcount = M;
  if constexpr (MOE) {
    mcount = counts[z];
    if (m0 >= mcount) return;
  }
  int mv = mcount - m0;
  if (mv > 128) mv = 128;

  int tid = threadIdx.x;
  if (tid < 128) {
    int i = m0 + (tid < mv ? tid : (mv - 1));
    if constexpr (MOE) {
      int slot = slot_list[z * N_TOK + i];
      o_rows[tid] = slot;
      a_rows[tid] = (MODE == 2) ? (slot >> 1) : slot;
    } else {
      a_rows[tid] = i;
      o_rows[tid] = i;
    }
  }
  __syncthreads();

  const unsigned short* Bp = Bt + (long)z * b_expert_stride + (long)n0 * ldb;
  const float* biasp = bias + (long)z * bias_stride + n0;

  int lane = tid & 31, wid = tid >> 5;
  int wm = wid >> 1, wn = wid & 1;
  int kh = lane >> 4, lc = lane & 15;
  int r_st = tid >> 1;             // staging row (A and Bt tiles, 128 each)
  int c_st = (tid & 1) * 16;       // staging column offset (halves)
  const unsigned short* arow_p = A + (long)a_rows[r_st] * lda + c_st;
  const unsigned short* brow_p = Bp + (long)r_st * ldb + c_st;

  v8f acc[2][4];
#pragma unroll
  for (int mi = 0; mi < 2; ++mi)
#pragma unroll
    for (int ni = 0; ni < 4; ++ni)
#pragma unroll
      for (int r = 0; r < 8; ++r) acc[mi][ni][r] = 0.f;

  const int nk = K / 32;
  uint4 a0, a1, b0, b1;
  // prologue: stage tile 0
  a0 = *(const uint4*)(arow_p);
  a1 = *(const uint4*)(arow_p + 8);
  b0 = *(const uint4*)(brow_p);
  b1 = *(const uint4*)(brow_p + 8);
  {
    unsigned short* pa = &As[0][r_st * 40 + c_st];
    *(uint4*)pa = a0; *(uint4*)(pa + 8) = a1;
    unsigned short* pb = &Bs[0][r_st * 40 + c_st];
    *(uint4*)pb = b0; *(uint4*)(pb + 8) = b1;
  }
  __syncthreads();

  for (int kb = 0; kb < nk; ++kb) {
    int buf = kb & 1;
    if (kb + 1 < nk) {  // issue next tile's global loads early
      long ko = (long)(kb + 1) * 32;
      a0 = *(const uint4*)(arow_p + ko);
      a1 = *(const uint4*)(arow_p + ko + 8);
      b0 = *(const uint4*)(brow_p + ko);
      b1 = *(const uint4*)(brow_p + ko + 8);
      if (kb + 2 < nk) {
        __builtin_prefetch(arow_p + ko + 32, 0, 0);
        __builtin_prefetch(brow_p + ko + 32, 0, 0);
      }
    }

    v16bf af[2], bfr[4];
#pragma unroll
    for (int mi = 0; mi < 2; ++mi)
      af[mi] = load_frag(&As[buf][(wm * 32 + mi * 16 + lc) * 40], kh);
#pragma unroll
    for (int ni = 0; ni < 4; ++ni)
      bfr[ni] = load_frag(&Bs[buf][(wn * 64 + ni * 16 + lc) * 40], kh);
#pragma unroll
    for (int mi = 0; mi < 2; ++mi)
#pragma unroll
      for (int ni = 0; ni < 4; ++ni)
        acc[mi][ni] = wmma_bf16(af[mi], bfr[ni], acc[mi][ni]);

    if (kb + 1 < nk) {  // store next tile into alternate buffer
      unsigned short* pa = &As[buf ^ 1][r_st * 40 + c_st];
      *(uint4*)pa = a0; *(uint4*)(pa + 8) = a1;
      unsigned short* pb = &Bs[buf ^ 1][r_st * 40 + c_st];
      *(uint4*)pb = b0; *(uint4*)(pb + 8) = b1;
    }
    __syncthreads();
  }

  // Epilogue (C layout: lanes 0-15 -> M=r, lanes 16-31 -> M=8+r)
  int hh = lane >> 4;
#pragma unroll
  for (int mi = 0; mi < 2; ++mi)
#pragma unroll
    for (int ni = 0; ni < 4; ++ni) {
      v8f a = acc[mi][ni];
#pragma unroll
      for (int r = 0; r < 8; ++r) {
        int ml = wm * 32 + mi * 16 + r + hh * 8;
        if (ml >= mv) continue;
        int n  = wn * 64 + ni * 16 + lc;
        int gn = n0 + n;
        long orow = o_rows[ml];
        float v = a[r];
        if constexpr (MODE == 0) {
          v += biasp[n];
          ((unsigned short*)Out)[orow * (long)ldo + gn] = f2bf(v);
        } else if constexpr (MODE == 1) {
          v += biasp[n] + Res[orow * (long)ldo + gn];
          ((float*)Out)[orow * (long)ldo + gn] = v;
        } else if constexpr (MODE == 2) {
          v += biasp[n];
          float ge = 0.5f * v * (1.0f + erff(v * 0.70710678118654752f));
          ((unsigned short*)Out)[orow * (long)ldo + gn] = f2bf(ge);
        } else if constexpr (MODE == 3) {
          float prev = first_chunk ? biasp[n]
                                   : ((float*)Out)[orow * (long)ldo + gn];
          ((float*)Out)[orow * (long)ldo + gn] = v + prev;
        } else {  // MODE 4: V stored transposed per head: [b*16+h][d][seq]
          v += biasp[n];
          long off = ((long)(orow >> 11) * 1024 + gn) * SEQ + (orow & (SEQ - 1));
          ((unsigned short*)Out)[off] = f2bf(v);
        }
      }
    }
}

// ---------------------------------------------------------------------------
// Flash attention, barrier-free: grid (S/64, B*H), 128 threads (4 waves).
// Each wave autonomously owns a 16-row Q strip. K fragments load straight
// from global; V is pre-transposed ([bh][d][seq]) so PV B-fragments also load
// straight from global. Only LDS use: per-wave P C-layout -> A-layout bounce.
// ---------------------------------------------------------------------------
__global__ __launch_bounds__(128)
void flash_attn_kernel(const unsigned short* __restrict__ q,
                       const unsigned short* __restrict__ k,
                       const unsigned short* __restrict__ vT,
                       unsigned short* __restrict__ ctx) {
  __shared__ __align__(16) unsigned short Ps[4 * 16 * 72];
  int tid = threadIdx.x, lane = tid & 31, wid = tid >> 5;
  int bh = blockIdx.y;
  int b = bh >> 4, h = bh & 15;
  long tb = (long)b * SEQ;
  int q0 = blockIdx.x * 64;
  int hh = lane >> 4, lc = lane & 15;

  const unsigned short* qrow =
      q + (tb + q0 + wid * 16 + lc) * (long)D_MODEL + h * D_HEAD;
  v16bf qf[2];
  qf[0] = load_frag(qrow, hh);
  qf[1] = load_frag(qrow + 32, hh);

  const unsigned short* vbase = vT + (long)bh * D_HEAD * SEQ;

  v8f ao[4];
  float mrun[8], lrun[8];
#pragma unroll
  for (int ni = 0; ni < 4; ++ni)
#pragma unroll
    for (int r = 0; r < 8; ++r) ao[ni][r] = 0.f;
#pragma unroll
  for (int r = 0; r < 8; ++r) { mrun[r] = -3e38f; lrun[r] = 0.f; }

  for (int kt = 0; kt < SEQ; kt += 64) {
    // S = Q @ K^T
    v8f as[4];
#pragma unroll
    for (int ni = 0; ni < 4; ++ni)
#pragma unroll
      for (int r = 0; r < 8; ++r) as[ni][r] = 0.f;
#pragma unroll
    for (int ks = 0; ks < 2; ++ks)
#pragma unroll
      for (int ni = 0; ni < 4; ++ni) {
        const unsigned short* krow =
            k + (tb + kt + ni * 16 + lc) * (long)D_MODEL + h * D_HEAD + ks * 32;
        v16bf bk = load_frag(krow, hh);
        as[ni] = wmma_bf16(qf[ks], bk, as[ni]);
      }
#pragma unroll
    for (int ni = 0; ni < 4; ++ni)
#pragma unroll
      for (int r = 0; r < 8; ++r) as[ni][r] *= 0.125f;  // 1/sqrt(64)

    // Online softmax (rows live in 16-lane halves: M = r + hh*8)
#pragma unroll
    for (int r = 0; r < 8; ++r) {
      float mx = fmaxf(fmaxf(as[0][r], as[1][r]), fmaxf(as[2][r], as[3][r]));
#pragma unroll
      for (int off = 8; off >= 1; off >>= 1)
        mx = fmaxf(mx, __shfl_xor(mx, off, 16));
      float mnew = fmaxf(mrun[r], mx);
      float corr = __expf(mrun[r] - mnew);
      mrun[r] = mnew;
      float rs = 0.f;
#pragma unroll
      for (int ni = 0; ni < 4; ++ni) {
        float pe = __expf(as[ni][r] - mnew);
        rs += pe;
        Ps[wid * 16 * 72 + (r + hh * 8) * 72 + ni * 16 + lc] = f2bf(pe);
      }
#pragma unroll
      for (int off = 8; off >= 1; off >>= 1) rs += __shfl_xor(rs, off, 16);
      lrun[r] = lrun[r] * corr + rs;
#pragma unroll
      for (int ni = 0; ni < 4; ++ni) ao[ni][r] *= corr;
    }

    // O += P @ V (P from per-wave LDS in A layout, V direct from global)
#pragma unroll
    for (int ks = 0; ks < 2; ++ks) {
      v16bf pf = load_frag(&Ps[wid * 16 * 72 + lc * 72 + ks * 32], hh);
#pragma unroll
      for (int ni = 0; ni < 4; ++ni) {
        v16bf vf =
            load_frag(vbase + (long)(ni * 16 + lc) * SEQ + kt + ks * 32, hh);
        ao[ni] = wmma_bf16(pf, vf, ao[ni]);
      }
    }
  }

#pragma unroll
  for (int ni = 0; ni < 4; ++ni)
#pragma unroll
    for (int r = 0; r < 8; ++r) {
      float val = ao[ni][r] / lrun[r];
      long row = tb + q0 + wid * 16 + r + hh * 8;
      ctx[row * (long)D_MODEL + h * D_HEAD + ni * 16 + lc] = f2bf(val);
    }
}

// ---------------------------------------------------------------------------
// Router: one thread per token. Lang softmax bias, top-2, expert slot lists.
// ---------------------------------------------------------------------------
__global__ __launch_bounds__(256)
void router_kernel(const unsigned short* __restrict__ xn,
                   const float* __restrict__ wr, const float* __restrict__ br,
                   const float* __restrict__ wl, const float* __restrict__ bl,
                   float* __restrict__ slot_w, int* __restrict__ counts,
                   int* __restrict__ lists) {
  int t = blockIdx.x * blockDim.x + threadIdx.x;
  if (t >= N_TOK) return;
  float rl[8], ll[4];
#pragma unroll
  for (int i = 0; i < 8; ++i) rl[i] = br[i];
#pragma unroll
  for (int i = 0; i < 4; ++i) ll[i] = bl[i];
  const unsigned short* xr = xn + (long)t * D_MODEL;
  for (int d = 0; d < D_MODEL; ++d) {
    float xv = bf2f(xr[d]);
    const float* wrr = wr + d * 8;
    const float* wlr = wl + d * 4;
#pragma unroll
    for (int i = 0; i < 8; ++i) rl[i] += xv * wrr[i];
#pragma unroll
    for (int i = 0; i < 4; ++i) ll[i] += xv * wlr[i];
  }
  float lm = fmaxf(fmaxf(ll[0], ll[1]), fmaxf(ll[2], ll[3]));
  float ls = 0.f, lp[4];
#pragma unroll
  for (int i = 0; i < 4; ++i) { lp[i] = __expf(ll[i] - lm); ls += lp[i]; }
#pragma unroll
  for (int i = 0; i < 4; ++i) lp[i] /= ls;
#pragma unroll
  for (int e = 0; e < 8; ++e) {
    int idx = e & 3;
    float ew = lp[idx] + (idx < 3 ? lp[idx + 1] : 0.f);
    rl[e] += 0.1f * ew;
  }
  float rm = rl[0];
#pragma unroll
  for (int i = 1; i < 8; ++i) rm = fmaxf(rm, rl[i]);
  float rs = 0.f, rp[8];
#pragma unroll
  for (int i = 0; i < 8; ++i) { rp[i] = __expf(rl[i] - rm); rs += rp[i]; }
#pragma unroll
  for (int i = 0; i < 8; ++i) rp[i] /= rs;
  int i1 = 0;
#pragma unroll
  for (int i = 1; i < 8; ++i) if (rp[i] > rp[i1]) i1 = i;
  float p1 = rp[i1];
  rp[i1] = -1.f;
  int i2 = 0;
#pragma unroll
  for (int i = 1; i < 8; ++i) if (rp[i] > rp[i2]) i2 = i;
  float p2 = rp[i2];
  float inv = 1.0f / (p1 + p2);
  slot_w[2 * t]     = p1 * inv;
  slot_w[2 * t + 1] = p2 * inv;
  int pos1 = atomicAdd(&counts[i1], 1);
  lists[i1 * N_TOK + pos1] = 2 * t;
  int pos2 = atomicAdd(&counts[i2], 1);
  lists[i2 * N_TOK + pos2] = 2 * t + 1;
}

// ---------------------------------------------------------------------------
// Final combine: out = x1 + w0*eo[2t] + w1*eo[2t+1]
// ---------------------------------------------------------------------------
__global__ __launch_bounds__(256)
void combine_kernel(const float* __restrict__ x1, const float* __restrict__ eo,
                    const float* __restrict__ sw, float* __restrict__ out) {
  long i = (long)blockIdx.x * 256 + threadIdx.x;
  long t = i >> 10;
  int  d = (int)(i & 1023);
  out[i] = x1[i] + sw[2 * t] * eo[(2 * t) * (long)D_MODEL + d] +
           sw[2 * t + 1] * eo[(2 * t + 1) * (long)D_MODEL + d];
}

// ---------------------------------------------------------------------------
// Host launcher
// ---------------------------------------------------------------------------
extern "C" void kernel_launch(void* const* d_in, const int* in_sizes, int n_in,
                              void* d_out, int out_size, void* d_ws,
                              size_t ws_size, hipStream_t stream) {
  const float* x       = (const float*)d_in[0];
  const float* ln1_g   = (const float*)d_in[1];
  const float* ln1_b   = (const float*)d_in[2];
  const float* ln2_g   = (const float*)d_in[3];
  const float* ln2_b   = (const float*)d_in[4];
  const float* wq      = (const float*)d_in[5];
  const float* bq      = (const float*)d_in[6];
  const float* wk      = (const float*)d_in[7];
  const float* bk      = (const float*)d_in[8];
  const float* wv      = (const float*)d_in[9];
  const float* bv      = (const float*)d_in[10];
  const float* wo      = (const float*)d_in[11];
  const float* bo      = (const float*)d_in[12];
  const float* w_rout  = (const float*)d_in[13];
  const float* b_rout  = (const float*)d_in[14];
  const float* w_lang  = (const float*)d_in[15];
  const float* b_lang  = (const float*)d_in[16];
  const float* w1      = (const float*)d_in[17];
  const float* b1      = (const float*)d_in[18];
  const float* w2      = (const float*)d_in[19];
  const float* b2      = (const float*)d_in[20];
  float* out = (float*)d_out;

  char* ws = (char*)d_ws;
  size_t cur = 0;
  auto take = [&](size_t bytes) -> char* {
    cur = (cur + 255) & ~(size_t)255;
    char* p = ws + cur;
    cur += bytes;
    return p;
  };
  const long DD = (long)D_MODEL * D_MODEL;       // 1M
  const long DF = (long)D_MODEL * EXPERT_SIZE;   // 4M
  unsigned short* wqT = (unsigned short*)take(DD * 2);
  unsigned short* wkT = (unsigned short*)take(DD * 2);
  unsigned short* wvT = (unsigned short*)take(DD * 2);
  unsigned short* woT = (unsigned short*)take(DD * 2);
  unsigned short* w1T = (unsigned short*)take((long)N_EXPERTS * DF * 2);
  unsigned short* w2T = (unsigned short*)take((long)N_EXPERTS * DF * 2);
  unsigned short* xn_bf = (unsigned short*)take((long)N_TOK * D_MODEL * 2);
  unsigned short* q_bf  = (unsigned short*)take((long)N_TOK * D_MODEL * 2);
  unsigned short* k_bf  = (unsigned short*)take((long)N_TOK * D_MODEL * 2);
  unsigned short* vT_bf = (unsigned short*)take((long)N_TOK * D_MODEL * 2);
  unsigned short* c_bf  = (unsigned short*)take((long)N_TOK * D_MODEL * 2);
  float* x1   = (float*)take((long)N_TOK * D_MODEL * 4);
  unsigned short* h_ck = (unsigned short*)take((long)N_SLOTS * F_CHUNK * 2);
  float* eo   = (float*)take((long)N_SLOTS * D_MODEL * 4);
  float* sw   = (float*)take((long)N_SLOTS * 4);
  int* counts = (int*)take(N_EXPERTS * 4);
  int* lists  = (int*)take((long)N_EXPERTS * N_TOK * 4);
  (void)ws_size; (void)n_in; (void)in_sizes; (void)out_size;

  // 1) Weight convert + transpose: w[K][N] f32 -> wT[N][K] bf16
  dim3 gT(D_MODEL / 32, D_MODEL / 32, 1);
  transpose_convert_kernel<<<gT, 256, 0, stream>>>(wq, wqT, D_MODEL, D_MODEL);
  transpose_convert_kernel<<<gT, 256, 0, stream>>>(wk, wkT, D_MODEL, D_MODEL);
  transpose_convert_kernel<<<gT, 256, 0, stream>>>(wv, wvT, D_MODEL, D_MODEL);
  transpose_convert_kernel<<<gT, 256, 0, stream>>>(wo, woT, D_MODEL, D_MODEL);
  dim3 gT1(EXPERT_SIZE / 32, D_MODEL / 32, N_EXPERTS);   // w1 [D][F] -> [F][D]
  transpose_convert_kernel<<<gT1, 256, 0, stream>>>(w1, w1T, D_MODEL, EXPERT_SIZE);
  dim3 gT2(D_MODEL / 32, EXPERT_SIZE / 32, N_EXPERTS);   // w2 [F][D] -> [D][F]
  transpose_convert_kernel<<<gT2, 256, 0, stream>>>(w2, w2T, EXPERT_SIZE, D_MODEL);
  hipMemsetAsync(counts, 0, N_EXPERTS * sizeof(int), stream);

  // 2) LN1 -> xn (bf16)
  layernorm_kernel<<<N_TOK, 256, 0, stream>>>(x, ln1_g, ln1_b, xn_bf);

  // 3) Q/K/V projections (WMMA). V written transposed per head.
  dim3 gQKV(D_MODEL / 128, N_TOK / 128, 1);
  gemm_bf16_kernel<0><<<gQKV, 256, 0, stream>>>(xn_bf, D_MODEL, wqT, D_MODEL,
      0, bq, 0, q_bf, D_MODEL, nullptr, N_TOK, D_MODEL, D_MODEL, nullptr,
      nullptr, 0);
  gemm_bf16_kernel<0><<<gQKV, 256, 0, stream>>>(xn_bf, D_MODEL, wkT, D_MODEL,
      0, bk, 0, k_bf, D_MODEL, nullptr, N_TOK, D_MODEL, D_MODEL, nullptr,
      nullptr, 0);
  gemm_bf16_kernel<4><<<gQKV, 256, 0, stream>>>(xn_bf, D_MODEL, wvT, D_MODEL,
      0, bv, 0, vT_bf, SEQ, nullptr, N_TOK, D_MODEL, D_MODEL, nullptr,
      nullptr, 0);

  // 4) Flash attention -> ctx (bf16), barrier-free
  dim3 gAtt(SEQ / 64, 4 * N_HEADS, 1);
  flash_attn_kernel<<<gAtt, 128, 0, stream>>>(q_bf, k_bf, vT_bf, c_bf);

  // 5) O projection + residual -> x1 (fp32)
  gemm_bf16_kernel<1><<<gQKV, 256, 0, stream>>>(c_bf, D_MODEL, woT, D_MODEL,
      0, bo, 0, x1, D_MODEL, x, N_TOK, D_MODEL, D_MODEL, nullptr, nullptr, 0);

  // 6) LN2 -> xn (bf16, buffer reuse)
  layernorm_kernel<<<N_TOK, 256, 0, stream>>>(x1, ln2_g, ln2_b, xn_bf);

  // 7) Router: top-2 experts per token, slot lists
  router_kernel<<<N_TOK / 256, 256, 0, stream>>>(xn_bf, w_rout, b_rout, w_lang,
                                                 b_lang, sw, counts, lists);

  // 8) Routed MoE in F chunks: GEMM1(+GeLU) then GEMM2 accumulation
  for (int c = 0; c < EXPERT_SIZE / F_CHUNK; ++c) {
    dim3 g1(F_CHUNK / 128, N_TOK / 128, N_EXPERTS);
    gemm_bf16_kernel<2><<<g1, 256, 0, stream>>>(
        xn_bf, D_MODEL, w1T + (long)c * F_CHUNK * D_MODEL, D_MODEL, DF,
        b1 + (long)c * F_CHUNK, EXPERT_SIZE, h_ck, F_CHUNK, nullptr, 0,
        F_CHUNK, D_MODEL, lists, counts, 0);
    dim3 g2(D_MODEL / 128, N_TOK / 128, N_EXPERTS);
    gemm_bf16_kernel<3><<<g2, 256, 0, stream>>>(
        h_ck, F_CHUNK, w2T + (long)c * F_CHUNK, EXPERT_SIZE, DF,
        b2, D_MODEL, eo, D_MODEL, nullptr, 0, D_MODEL, F_CHUNK, lists, counts,
        (c == 0) ? 1 : 0);
  }

  // 9) Combine: out = x1 + w0*eo0 + w1*eo1
  combine_kernel<<<(N_TOK * D_MODEL) / 256, 256, 0, stream>>>(x1, eo, sw, out);
}
